// AttentionLSTMDecoder_36275293782217
// MI455X (gfx1250) — matrix operand order repaired and checked by reading, not compiled, not run on hardware
//
#include <hip/hip_runtime.h>
#include <hip/hip_bf16.h>
#include <math.h>

// ---------------------------------------------------------------------------
// Model dims
// ---------------------------------------------------------------------------
#define HD   768
#define NH   16
#define DH   48
#define NL   3
#define BB   64
#define SS   32
#define VV   32000
#define P49  49
#define CH   2048
#define EDIM 512
#define FUSK (EDIM + HD)   // 1280
#define G4   (4 * HD)      // 3072
#define H2   (2 * HD)      // 1536

typedef __attribute__((ext_vector_type(16))) __bf16 bf16x16;
typedef __attribute__((ext_vector_type(8)))  __bf16 bf16x8;
typedef __attribute__((ext_vector_type(8)))  float  f32x8;

// round-to-nearest-even float -> bf16 (storage-only conversion)
__device__ __forceinline__ __bf16 f2bf(float f) {
    union { float f; unsigned u; } v; v.f = f;
    unsigned r = v.u + 0x7FFFu + ((v.u >> 16) & 1u);
    unsigned short h = (unsigned short)(r >> 16);
    union { unsigned short s; __bf16 b; } o; o.s = h;
    return o.b;
}

__device__ __forceinline__ float sigmf(float x) { return 1.0f / (1.0f + expf(-x)); }

// ---------------------------------------------------------------------------
// fp32 -> bf16 conversion (weights once per launch; h-state mirrors)
// ---------------------------------------------------------------------------
__global__ void k_f32_to_bf16(const float* __restrict__ src,
                              __bf16* __restrict__ dst, long long n) {
    long long i = (long long)blockIdx.x * blockDim.x + threadIdx.x;
    long long stride = (long long)gridDim.x * blockDim.x;
    for (; i < n; i += stride) dst[i] = f2bf(src[i]);
}

// ---------------------------------------------------------------------------
// One software-pipelined K-pass of the WMMA GEMM: 4 M-tile accumulators,
// B fragment shared across the 4 WMMAs; next iteration's fragments are
// loaded before the current iteration's WMMAs issue so loads overlap the
// matrix pipe instead of s_wait_loadcnt 0 stalling before every WMMA.
// ---------------------------------------------------------------------------
__device__ __forceinline__ void gemm_pass(const __bf16* __restrict__ aBase,
                                          const __bf16* __restrict__ wRow,
                                          int K, int hi, f32x8 acc[4]) {
    // current fragments (k)
    bf16x16 bc = *(const bf16x16*)(wRow + hi * 16);
    bf16x8 a0c[4], a1c[4];
    #pragma unroll
    for (int m = 0; m < 4; ++m) {
        const __bf16* ap = aBase + (size_t)m * 16 * K + hi * 16;
        a0c[m] = *(const bf16x8*)ap;
        a1c[m] = *(const bf16x8*)(ap + 8);
    }

    int k = 0;
    for (; k + 32 < K; k += 32) {
        // prefetch upcoming weight cachelines (global_prefetch_b8)
        __builtin_prefetch(wRow + k + 544, 0, 1);

        // next fragments (k+32) issued BEFORE current WMMAs
        const bf16x16 bn = *(const bf16x16*)(wRow + (k + 32) + hi * 16);
        bf16x8 a0n[4], a1n[4];
        #pragma unroll
        for (int m = 0; m < 4; ++m) {
            const __bf16* ap = aBase + (size_t)m * 16 * K + (k + 32) + hi * 16;
            a0n[m] = *(const bf16x8*)ap;
            a1n[m] = *(const bf16x8*)(ap + 8);
        }

        #pragma unroll
        for (int m = 0; m < 4; ++m) {
            const bf16x16 afrag = __builtin_shufflevector(
                a0c[m], a1c[m], 0, 1, 2, 3, 4, 5, 6, 7, 8, 9, 10, 11, 12, 13, 14, 15);
            acc[m] = __builtin_amdgcn_wmma_f32_16x16x32_bf16(
                         false, afrag, false, bc, (short)0, acc[m], false, false);
        }

        bc = bn;
        #pragma unroll
        for (int m = 0; m < 4; ++m) { a0c[m] = a0n[m]; a1c[m] = a1n[m]; }
    }

    // drain
    #pragma unroll
    for (int m = 0; m < 4; ++m) {
        const bf16x16 afrag = __builtin_shufflevector(
            a0c[m], a1c[m], 0, 1, 2, 3, 4, 5, 6, 7, 8, 9, 10, 11, 12, 13, 14, 15);
        acc[m] = __builtin_amdgcn_wmma_f32_16x16x32_bf16(
                     false, afrag, false, bc, (short)0, acc[m], false, false);
    }
}

// ---------------------------------------------------------------------------
// WMMA bf16 GEMM: C[M,N] = A@W^T (+ A2@W2^T) + bias (+bias2)
// Each wave computes a 64-row x 16-col strip (4 accumulators). Optional
// second (A2,W2) pass fuses the LSTM ih+hh gate GEMMs without a C round-trip.
// blockDim = (32,8); grid = (ceil(N/16/8), M/64). M must be a multiple of 64.
// Fragment k order: contiguous, k = 16*(lane/16) + s for both A and B (wave32).
// ---------------------------------------------------------------------------
__global__ void k_gemm_wmma(const __bf16* __restrict__ A,
                            const __bf16* __restrict__ W,
                            const __bf16* __restrict__ A2,   // optional
                            const __bf16* __restrict__ W2,   // optional
                            const float* __restrict__ bias,
                            const float* __restrict__ bias2, // optional
                            float* __restrict__ Cf,          // fp32 out (optional)
                            __bf16* __restrict__ Cb,         // bf16 out (optional)
                            int N, int K, long long ldc) {
    const int lane  = threadIdx.x;          // 0..31
    const int wave  = threadIdx.y;          // 0..7
    const int tileN = blockIdx.x * blockDim.y + wave;
    if (tileN * 16 >= N) return;            // wave-uniform: EXEC stays full

    const int hi    = lane >> 4;
    const int l15   = lane & 15;
    const int mbase = blockIdx.y * 64;
    const int colW  = tileN * 16 + l15;

    f32x8 acc[4];
    #pragma unroll
    for (int m = 0; m < 4; ++m)
        acc[m] = (f32x8){0.f, 0.f, 0.f, 0.f, 0.f, 0.f, 0.f, 0.f};

    gemm_pass(A + (size_t)(mbase + l15) * K, W + (size_t)colW * K, K, hi, acc);
    if (A2)
        gemm_pass(A2 + (size_t)(mbase + l15) * K, W2 + (size_t)colW * K, K, hi, acc);

    float bv = bias ? bias[colW] : 0.0f;
    if (bias2) bv += bias2[colW];
    #pragma unroll
    for (int m = 0; m < 4; ++m)
        #pragma unroll
        for (int r = 0; r < 8; ++r) {
            const float v = acc[m][r] + bv;
            const size_t idx = (size_t)(mbase + m * 16 + r + hi * 8) * ldc + colW;
            if (Cf) Cf[idx] = v;
            if (Cb) Cb[idx] = f2bf(v);
        }
}

// ---------------------------------------------------------------------------
// LayerNorm over last dim D; act 0=none 1=relu 2=tanh.
// Writes fp32 and/or bf16 outputs (either pointer may be null).
// ---------------------------------------------------------------------------
__global__ void k_layernorm(const float* __restrict__ X, const float* __restrict__ lnp,
                            float* __restrict__ Yf, __bf16* __restrict__ Yb,
                            int D, long long ystride, int act) {
    const int row = blockIdx.x;
    const int tid = threadIdx.x;
    const float* x = X + (size_t)row * D;
    __shared__ float red[256];

    float s = 0.f;
    for (int i = tid; i < D; i += 256) s += x[i];
    red[tid] = s; __syncthreads();
    for (int st = 128; st > 0; st >>= 1) {
        if (tid < st) red[tid] += red[tid + st];
        __syncthreads();
    }
    const float mu = red[0] / (float)D;
    __syncthreads();

    float v = 0.f;
    for (int i = tid; i < D; i += 256) { float d = x[i] - mu; v += d * d; }
    red[tid] = v; __syncthreads();
    for (int st = 128; st > 0; st >>= 1) {
        if (tid < st) red[tid] += red[tid + st];
        __syncthreads();
    }
    const float rs = rsqrtf(red[0] / (float)D + 1e-5f);

    for (int i = tid; i < D; i += 256) {
        float y = (x[i] - mu) * rs * lnp[i] + lnp[D + i];
        if (act == 1)      y = fmaxf(y, 0.f);
        else if (act == 2) y = tanhf(y);
        const size_t idx = (size_t)row * ystride + i;
        if (Yf) Yf[idx] = y;
        if (Yb) Yb[idx] = f2bf(y);
    }
}

// ---------------------------------------------------------------------------
// Attention for one timestep: scores -> softmax(49) -> context (bf16 out)
// grid = B*NH blocks, 64 threads. Q,K,V h-major inside HD (h*DH+d).
// ---------------------------------------------------------------------------
__global__ void k_attention(const float* __restrict__ Q,
                            const float* __restrict__ Kf,
                            const float* __restrict__ Vf,
                            __bf16* __restrict__ ctx) {
    const int b = blockIdx.x / NH;
    const int h = blockIdx.x % NH;
    const int tid = threadIdx.x;
    __shared__ float q[DH];
    __shared__ float sc[P49];

    if (tid < DH) q[tid] = Q[(size_t)b * HD + h * DH + tid];
    __syncthreads();

    if (tid < P49) {
        const float* kp = Kf + ((size_t)(b * P49 + tid) * HD + h * DH);
        float s = 0.f;
        #pragma unroll
        for (int d = 0; d < DH; ++d) s += q[d] * kp[d];
        sc[tid] = s * 0.14433756729740643f;  // 1/sqrt(48)
    }
    __syncthreads();

    if (tid == 0) {
        float m = sc[0];
        for (int k = 1; k < P49; ++k) m = fmaxf(m, sc[k]);
        float sum = 0.f;
        for (int k = 0; k < P49; ++k) { float e = expf(sc[k] - m); sc[k] = e; sum += e; }
        float inv = 1.f / sum;
        for (int k = 0; k < P49; ++k) sc[k] *= inv;
    }
    __syncthreads();

    if (tid < DH) {
        float acc = 0.f;
        for (int k = 0; k < P49; ++k)
            acc += sc[k] * Vf[(size_t)(b * P49 + k) * HD + h * DH + tid];
        ctx[(size_t)b * HD + h * DH + tid] = f2bf(acc);
    }
}

// ---------------------------------------------------------------------------
// LSTM cell (torch gate order i,f,g,o); updates fp32 h,c and bf16 h-mirror.
// ---------------------------------------------------------------------------
__global__ void k_lstm_cell(const float* __restrict__ g,
                            float* __restrict__ h, float* __restrict__ c,
                            __bf16* __restrict__ hb, int n) {
    int idx = blockIdx.x * blockDim.x + threadIdx.x;
    if (idx >= n) return;
    const int b = idx / HD, j = idx % HD;
    const size_t base = (size_t)b * G4;
    const float gi = g[base + j];
    const float gf = g[base + HD + j];
    const float gg = g[base + 2 * HD + j];
    const float go = g[base + 3 * HD + j];
    const float cn = sigmf(gf) * c[idx] + sigmf(gi) * tanhf(gg);
    const float hn = sigmf(go) * tanhf(cn);
    c[idx]  = cn;
    h[idx]  = hn;
    hb[idx] = f2bf(hn);
}

// ---------------------------------------------------------------------------
// small data-movement kernels
// ---------------------------------------------------------------------------
__global__ void k_copy(const float* __restrict__ src, float* __restrict__ dst, long long n) {
    long long i = (long long)blockIdx.x * blockDim.x + threadIdx.x;
    long long stride = (long long)gridDim.x * blockDim.x;
    for (; i < n; i += stride) dst[i] = src[i];
}

// bf16 strided pack: dst[r*dstStride + c] = src[r*srcStride + c]
__global__ void k_pack_bf(const __bf16* __restrict__ src, __bf16* __restrict__ dst,
                          int rows, int cols, long long srcStride, long long dstStride) {
    long long n = (long long)rows * cols;
    long long i = (long long)blockIdx.x * blockDim.x + threadIdx.x;
    long long stride = (long long)gridDim.x * blockDim.x;
    for (; i < n; i += stride) {
        int r = (int)(i / cols), cl = (int)(i % cols);
        dst[(size_t)r * dstStride + cl] = src[(size_t)r * srcStride + cl];
    }
}

// attn_feat (B,2048,49) fp32 -> feat (B*49, 2048) bf16
__global__ void k_feat_transpose(const float* __restrict__ af, __bf16* __restrict__ feat) {
    long long n = (long long)BB * P49 * CH;
    long long i = (long long)blockIdx.x * blockDim.x + threadIdx.x;
    long long stride = (long long)gridDim.x * blockDim.x;
    for (; i < n; i += stride) {
        int ch = (int)(i % CH);
        int p  = (int)((i / CH) % P49);
        int b  = (int)(i / ((long long)CH * P49));
        feat[i] = f2bf(af[((size_t)b * CH + ch) * P49 + p]);
    }
}

// embeds stored (S,B,512) bf16 so each timestep slice is contiguous
__global__ void k_embed_gather(const int* __restrict__ ids, const float* __restrict__ emb,
                               __bf16* __restrict__ out) {
    long long n = (long long)BB * SS * EDIM;
    long long i = (long long)blockIdx.x * blockDim.x + threadIdx.x;
    long long stride = (long long)gridDim.x * blockDim.x;
    for (; i < n; i += stride) {
        int d = (int)(i % EDIM);
        int b = (int)((i / EDIM) % BB);
        int s = (int)(i / ((long long)EDIM * BB));
        out[i] = f2bf(emb[(size_t)ids[b * SS + s] * EDIM + d]);
    }
}

// ---------------------------------------------------------------------------
// host-side orchestration
// ---------------------------------------------------------------------------
extern "C" void kernel_launch(void* const* d_in, const int* in_sizes, int n_in,
                              void* d_out, int out_size, void* d_ws, size_t ws_size,
                              hipStream_t stream) {
    (void)in_sizes; (void)n_in; (void)out_size; (void)ws_size;

    const int*   word_ids  = (const int*)  d_in[0];
    const float* cnn_feat  = (const float*)d_in[1];
    const float* attn_feat = (const float*)d_in[2];
    const float* emb       = (const float*)d_in[3];
    const float* attn_f_w  = (const float*)d_in[4];
    const float* attn_f_b  = (const float*)d_in[5];
    const float* attn_qkv_w= (const float*)d_in[6];
    const float* attn_qkv_b= (const float*)d_in[7];
    const float* attn_o_w  = (const float*)d_in[8];
    const float* attn_o_b  = (const float*)d_in[9];
    const float* attn_ln   = (const float*)d_in[10];
    const float* fus_w1    = (const float*)d_in[11];
    const float* fus_b1    = (const float*)d_in[12];
    const float* fus_ln1   = (const float*)d_in[13];
    const float* fus_w2    = (const float*)d_in[14];
    const float* fus_b2    = (const float*)d_in[15];
    const float* fus_ln2   = (const float*)d_in[16];
    const float* lstm_wih  = (const float*)d_in[17];
    const float* lstm_whh  = (const float*)d_in[18];
    const float* lstm_bih  = (const float*)d_in[19];
    const float* lstm_bhh  = (const float*)d_in[20];
    const float* hinit_w1  = (const float*)d_in[21];
    const float* hinit_b1  = (const float*)d_in[22];
    const float* hinit_ln1 = (const float*)d_in[23];
    const float* hinit_w2  = (const float*)d_in[24];
    const float* hinit_b2  = (const float*)d_in[25];
    const float* hinit_ln2 = (const float*)d_in[26];
    const float* cinit_w1  = (const float*)d_in[27];
    const float* cinit_b1  = (const float*)d_in[28];
    const float* cinit_ln1 = (const float*)d_in[29];
    const float* cinit_w2  = (const float*)d_in[30];
    const float* cinit_b2  = (const float*)d_in[31];
    const float* cinit_ln2 = (const float*)d_in[32];
    const float* out_w1    = (const float*)d_in[33];
    const float* out_b1    = (const float*)d_in[34];
    const float* out_ln1   = (const float*)d_in[35];
    const float* out_w2    = (const float*)d_in[36];
    const float* out_b2    = (const float*)d_in[37];
    const float* out_ln2   = (const float*)d_in[38];
    const float* out_w3    = (const float*)d_in[39];
    const float* out_b3    = (const float*)d_in[40];

    float* out = (float*)d_out;

    // ---- workspace allocator ----
    char* base = (char*)d_ws;
    size_t off = 0;
    auto alloc = [&](size_t bytes) -> void* {
        void* p = base + off;
        off += (bytes + 255) & ~(size_t)255;
        return p;
    };

    // bf16 weights
    __bf16* wf    = (__bf16*)alloc((size_t)HD * CH * 2);
    __bf16* wqkv  = (__bf16*)alloc((size_t)3 * HD * HD * 2);
    __bf16* wo    = (__bf16*)alloc((size_t)HD * HD * 2);
    __bf16* wfus1 = (__bf16*)alloc((size_t)HD * FUSK * 2);
    __bf16* wfus2 = (__bf16*)alloc((size_t)HD * HD * 2);
    __bf16* wih   = (__bf16*)alloc((size_t)NL * G4 * HD * 2);
    __bf16* whh   = (__bf16*)alloc((size_t)NL * G4 * HD * 2);
    __bf16* whi1  = (__bf16*)alloc((size_t)H2 * HD * 2);
    __bf16* whi2  = (__bf16*)alloc((size_t)HD * H2 * 2);
    __bf16* wci1  = (__bf16*)alloc((size_t)H2 * HD * 2);
    __bf16* wci2  = (__bf16*)alloc((size_t)HD * H2 * 2);
    __bf16* wo1   = (__bf16*)alloc((size_t)H2 * HD * 2);
    __bf16* wo2   = (__bf16*)alloc((size_t)HD * H2 * 2);
    __bf16* wv3   = (__bf16*)alloc((size_t)VV * HD * 2);

    // bf16 activations (GEMM inputs)
    __bf16* featb  = (__bf16*)alloc((size_t)BB * P49 * CH * 2);
    __bf16* fpbf   = (__bf16*)alloc((size_t)BB * P49 * HD * 2);
    __bf16* embeds = (__bf16*)alloc((size_t)SS * BB * EDIM * 2);
    __bf16* cnnbf  = (__bf16*)alloc((size_t)BB * HD * 2);
    __bf16* hbf    = (__bf16*)alloc((size_t)NL * BB * HD * 2);
    __bf16* ctxbf  = (__bf16*)alloc((size_t)BB * HD * 2);
    __bf16* xcatbf = (__bf16*)alloc((size_t)BB * FUSK * 2);
    __bf16* t768bf = (__bf16*)alloc((size_t)BB * HD * 2);
    __bf16* x2bf   = (__bf16*)alloc((size_t)BB * HD * 2);
    __bf16* y1bf   = (__bf16*)alloc((size_t)BB * H2 * 2);
    __bf16* y2bf   = (__bf16*)alloc((size_t)BB * HD * 2);

    // fp32 buffers
    float* Kb    = (float*)alloc((size_t)BB * P49 * HD * 4);
    float* Vb    = (float*)alloc((size_t)BB * P49 * HD * 4);
    float* hbuf  = (float*)alloc((size_t)NL * BB * HD * 4);
    float* cbuf  = (float*)alloc((size_t)NL * BB * HD * 4);
    float* qb    = (float*)alloc((size_t)BB * HD * 4);
    float* ctmp  = (float*)alloc((size_t)BB * HD * 4);
    float* t768a = (float*)alloc((size_t)BB * HD * 4);
    float* t1536 = (float*)alloc((size_t)BB * H2 * 4);
    float* gbuf  = (float*)alloc((size_t)BB * G4 * 4);
    float* h0    = (float*)alloc((size_t)BB * HD * 4);
    float* c0    = (float*)alloc((size_t)BB * HD * 4);

    auto conv = [&](const float* src, __bf16* dst, long long n) {
        int blocks = (int)(((n + 255) / 256) < 4096 ? ((n + 255) / 256) : 4096);
        k_f32_to_bf16<<<blocks, 256, 0, stream>>>(src, dst, n);
    };
    auto gemm2 = [&](const __bf16* A, const __bf16* W,
                     const __bf16* A2, const __bf16* W2,
                     const float* bias, const float* bias2,
                     float* Cf, __bf16* Cb, int M, int N, int K, long long ldc) {
        dim3 block(32, 8);
        dim3 grid((unsigned)((N / 16 + 7) / 8), (unsigned)(M / 64));
        k_gemm_wmma<<<grid, block, 0, stream>>>(A, W, A2, W2, bias, bias2,
                                                Cf, Cb, N, K, ldc);
    };
    auto gemm = [&](const __bf16* A, const __bf16* W, const float* bias,
                    float* Cf, __bf16* Cb, int M, int N, int K, long long ldc) {
        gemm2(A, W, nullptr, nullptr, bias, nullptr, Cf, Cb, M, N, K, ldc);
    };
    auto lnorm = [&](const float* X, const float* lnp, float* Yf, __bf16* Yb,
                     int rows, int D, long long ystride, int act) {
        k_layernorm<<<rows, 256, 0, stream>>>(X, lnp, Yf, Yb, D, ystride, act);
    };

    // ---- weight + input conversion ----
    conv(attn_f_w, wf,    (long long)HD * CH);
    conv(attn_qkv_w, wqkv,(long long)3 * HD * HD);
    conv(attn_o_w, wo,    (long long)HD * HD);
    conv(fus_w1, wfus1,   (long long)HD * FUSK);
    conv(fus_w2, wfus2,   (long long)HD * HD);
    conv(lstm_wih, wih,   (long long)NL * G4 * HD);
    conv(lstm_whh, whh,   (long long)NL * G4 * HD);
    conv(hinit_w1, whi1,  (long long)H2 * HD);
    conv(hinit_w2, whi2,  (long long)HD * H2);
    conv(cinit_w1, wci1,  (long long)H2 * HD);
    conv(cinit_w2, wci2,  (long long)HD * H2);
    conv(out_w1, wo1,     (long long)H2 * HD);
    conv(out_w2, wo2,     (long long)HD * H2);
    conv(out_w3, wv3,     (long long)VV * HD);
    conv(cnn_feat, cnnbf, (long long)BB * HD);

    const __bf16* Wq = wqkv;
    const __bf16* Wk = wqkv + (size_t)HD * HD;
    const __bf16* Wv = wqkv + (size_t)2 * HD * HD;
    const float* bq = attn_qkv_b;
    const float* bk = attn_qkv_b + HD;
    const float* bv = attn_qkv_b + 2 * HD;

    // ---- precompute: feat transpose, feature proj, K, V, embeddings ----
    k_feat_transpose<<<4096, 256, 0, stream>>>(attn_feat, featb);
    k_embed_gather<<<2048, 256, 0, stream>>>(word_ids, emb, embeds);
    gemm(featb, wf, attn_f_b, nullptr, fpbf, BB * P49, HD, CH, HD);  // bf16 out
    gemm(fpbf, Wk, bk, Kb, nullptr, BB * P49, HD, HD, HD);
    gemm(fpbf, Wv, bv, Vb, nullptr, BB * P49, HD, HD, HD);

    // ---- init MLPs -> h0, c0, broadcast to NL layers (fp32 + bf16 mirror) ----
    gemm(cnnbf, whi1, hinit_b1, t1536, nullptr, BB, H2, HD, H2);
    lnorm(t1536, hinit_ln1, nullptr, y1bf, BB, H2, H2, 1);
    gemm(y1bf, whi2, hinit_b2, t768a, nullptr, BB, HD, H2, HD);
    lnorm(t768a, hinit_ln2, h0, nullptr, BB, HD, HD, 2);

    gemm(cnnbf, wci1, cinit_b1, t1536, nullptr, BB, H2, HD, H2);
    lnorm(t1536, cinit_ln1, nullptr, y1bf, BB, H2, H2, 1);
    gemm(y1bf, wci2, cinit_b2, t768a, nullptr, BB, HD, H2, HD);
    lnorm(t768a, cinit_ln2, c0, nullptr, BB, HD, HD, 2);

    const long long nBH = (long long)BB * HD;
    for (int l = 0; l < NL; ++l) {
        k_copy<<<192, 256, 0, stream>>>(h0, hbuf + (size_t)l * nBH, nBH);
        k_copy<<<192, 256, 0, stream>>>(c0, cbuf + (size_t)l * nBH, nBH);
        conv(h0, hbf + (size_t)l * nBH, nBH);
    }

    // ---- timestep loop ----
    for (int t = 0; t < SS; ++t) {
        // q from top-layer hidden state (bf16 mirror)
        gemm(hbf + (size_t)(NL - 1) * nBH, Wq, bq, qb, nullptr, BB, HD, HD, HD);
        // attention: scores -> softmax -> context (bf16)
        k_attention<<<BB * NH, 64, 0, stream>>>(qb, Kb, Vb, ctxbf);
        // output projection + LN, write bf16 into xcat[:, 512:1280]
        gemm(ctxbf, wo, attn_o_b, ctmp, nullptr, BB, HD, HD, HD);
        lnorm(ctmp, attn_ln, nullptr, xcatbf + EDIM, BB, HD, FUSK, 0);
        // e_t into xcat[:, 0:512]
        k_pack_bf<<<128, 256, 0, stream>>>(embeds + (size_t)t * BB * EDIM, xcatbf,
                                           BB, EDIM, EDIM, FUSK);
        // fusion MLP
        gemm(xcatbf, wfus1, fus_b1, t768a, nullptr, BB, HD, FUSK, HD);
        lnorm(t768a, fus_ln1, nullptr, t768bf, BB, HD, HD, 1);
        gemm(t768bf, wfus2, fus_b2, t768a, nullptr, BB, HD, HD, HD);
        lnorm(t768a, fus_ln2, nullptr, x2bf, BB, HD, HD, 1);

        // 3-layer LSTM; ih and hh GEMMs fused in one kernel (no C round-trip)
        const __bf16* inp = x2bf;
        for (int l = 0; l < NL; ++l) {
            gemm2(inp, wih + (size_t)l * G4 * HD,
                  hbf + (size_t)l * nBH, whh + (size_t)l * G4 * HD,
                  lstm_bih + (size_t)l * G4, lstm_bhh + (size_t)l * G4,
                  gbuf, nullptr, BB, G4, HD, G4);
            k_lstm_cell<<<(BB * HD + 255) / 256, 256, 0, stream>>>(
                gbuf, hbuf + (size_t)l * nBH, cbuf + (size_t)l * nBH,
                hbf + (size_t)l * nBH, BB * HD);
            inp = hbf + (size_t)l * nBH;
        }

        // output head; logits written straight into d_out (B,S,V) at column t
        gemm(inp, wo1, out_b1, t1536, nullptr, BB, H2, HD, H2);
        lnorm(t1536, out_ln1, nullptr, y1bf, BB, H2, H2, 1);
        gemm(y1bf, wo2, out_b2, t768a, nullptr, BB, HD, H2, HD);
        lnorm(t768a, out_ln2, nullptr, y2bf, BB, HD, HD, 1);
        gemm(y2bf, wv3, out_b3, out + (size_t)t * VV, nullptr, BB, VV, HD,
             (long long)SS * VV);
    }

    // ---- final hidden/cell states appended after ys ----
    const long long ysN = (long long)BB * SS * VV;
    k_copy<<<576, 256, 0, stream>>>(hbuf, out + ysN, (long long)NL * BB * HD);
    k_copy<<<576, 256, 0, stream>>>(cbuf, out + ysN + (long long)NL * BB * HD,
                                    (long long)NL * BB * HD);
}
